// TTConv_66546223284784
// MI455X (gfx1250) — compile-verified, matrix operand order checked by reading, not run on hardware
//
#include <hip/hip_runtime.h>

typedef __attribute__((ext_vector_type(16))) __bf16 v16bf;
typedef __attribute__((ext_vector_type(8)))  float  v8f;

// builtin's operand type: int4 as a GCC vector, in global (1) / LDS (3) AS
typedef int v4i_vec __attribute__((vector_size(16)));
typedef __attribute__((address_space(1))) v4i_vec gv4i;
typedef __attribute__((address_space(3))) v4i_vec lv4i;

// ---- CDNA5 async global->LDS copy (ASYNCcnt-tracked), with fallback ----
#if __has_builtin(__builtin_amdgcn_global_load_async_to_lds_b128)
#define HAVE_ASYNC 1
#define ASYNC_CP128(g, l, off) \
    __builtin_amdgcn_global_load_async_to_lds_b128((gv4i*)(g), (lv4i*)(l), (off), 0)
#else
#define HAVE_ASYNC 0
#define ASYNC_CP128(g, l, off) \
    do { *(uint4*)((char*)(l) + (off)) = *(const uint4*)((const char*)(g) + (off)); } while (0)
#endif

__device__ __forceinline__ void wait_async0() {
#if HAVE_ASYNC
#if __has_builtin(__builtin_amdgcn_s_wait_asynccnt)
    __builtin_amdgcn_s_wait_asynccnt(0);
#else
    asm volatile("s_wait_asynccnt 0x0" ::: "memory");
#endif
#endif
}

// ---- workspace layout (bytes) ----
#define WS_XT    0u           // Xt  : bf16 [8][66][66][256] padded = 17,842,176 B
#define WS_BMAT  17842176u    // Bmat: bf16 [9][256][256]           =  1,179,648 B
#define WS_BIAS  19021824u    // bias2: f32 [256]                   =      1,024 B
#define WS_H1    19022848u    // H1  : f32 [4][8][16][16][4][8]     =  1,048,576 B
#define WS_W2    20071424u    // W2  : f32 [256][256][16]           =  4,194,304 B
// total ~23.2 MB

// H1[a,c,v,r,i,j] = sum_u G1[a,u,r,i] * G2[c,v,u,j]
__global__ __launch_bounds__(256) void k_prep1(const float* __restrict__ core1,
                                               const float* __restrict__ core2,
                                               float* __restrict__ H1) {
    int e = blockIdx.x * 256 + threadIdx.x;      // 262144 elems
    int j = e & 7, i = (e >> 3) & 3, r = (e >> 5) & 15;
    int v = (e >> 9) & 15, c = (e >> 13) & 7, a = (e >> 16) & 3;
    float s = 0.f;
    for (int u = 0; u < 16; ++u)
        s += core1[(a * 16 + u) * 64 + r * 4 + i] *
             core2[(c * 16 + v) * 128 + u * 8 + j];
    H1[e] = s;
}

// W2[o=(a,c,d)][cin=(i,j,k)][r] = sum_v H1[a,c,v,r,i,j] * G3[d,v,k]
__global__ __launch_bounds__(256) void k_prep2(const float* __restrict__ H1,
                                               const float* __restrict__ core3,
                                               float* __restrict__ W2) {
    int e = blockIdx.x * 256 + threadIdx.x;      // 1,048,576 elems
    int r = e & 15, k = (e >> 4) & 7, j = (e >> 7) & 7, i = (e >> 10) & 3;
    int d = (e >> 12) & 7, cc = (e >> 15) & 7, a = (e >> 18) & 3;
    float s = 0.f;
    for (int v = 0; v < 16; ++v)
        s += H1[((((a * 8 + cc) * 16 + v) * 16 + r) * 4 + i) * 8 + j] *
             core3[d * 128 + v * 8 + k];
    int o   = (a * 8 + cc) * 8 + d;
    int cin = (i * 8 + j) * 8 + k;
    W2[(o * 256 + cin) * 16 + r] = s;
}

// Bmat[t][o][cin] = bf16( sum_r W2[o][cin][r] * conv_w[r][t] )
__global__ __launch_bounds__(256) void k_prep3(const float* __restrict__ W2,
                                               const float* __restrict__ conv_w,
                                               __bf16* __restrict__ Bmat) {
    int e = blockIdx.x * 256 + threadIdx.x;      // 589,824 elems
    int cin = e & 255, o = (e >> 8) & 255, t = e >> 16;
    float s = 0.f;
    for (int r = 0; r < 16; ++r)
        s += W2[(o * 256 + cin) * 16 + r] * conv_w[r * 9 + t];
    Bmat[((size_t)t * 256 + o) * 256 + cin] = (__bf16)s;
}

// bias2[o] = sum_{cin,r} W2[o][cin][r] * conv_b[r]
__global__ __launch_bounds__(256) void k_bias(const float* __restrict__ W2,
                                              const float* __restrict__ conv_b,
                                              float* __restrict__ bias2) {
    int o = threadIdx.x;
    float s = 0.f;
    for (int cin = 0; cin < 256; ++cin)
        for (int r = 0; r < 16; ++r)
            s += W2[(o * 256 + cin) * 16 + r] * conv_b[r];
    bias2[o] = s;
}

// Xt[b][gy][gx][c] = bf16(X[b][c][gy-1][gx-1]) with zero halo (66x66 padded)
__global__ __launch_bounds__(256) void k_xT(const float* __restrict__ X,
                                            __bf16* __restrict__ Xt) {
    size_t e = (size_t)blockIdx.x * 256 + threadIdx.x;  // 8*66*66*256 = 8,921,088
    int ci = (int)(e & 255);
    int t  = (int)(e >> 8);
    int gx = t % 66;
    int t2 = t / 66;
    int gy = t2 % 66;
    int b  = t2 / 66;
    float v = 0.f;
    if (gx >= 1 && gx <= 64 && gy >= 1 && gy <= 64)
        v = X[(((size_t)b * 256 + ci) * 64 + (gy - 1)) * 64 + (gx - 1)];
    Xt[e] = (__bf16)v;
}

// Main fused conv-as-GEMM kernel: async-LDS double buffering + WMMA.
// M = 64 pixels/WG (4 rows x 16 cols), N = 256 out ch, K = 9 taps * 8 ch-blocks.
__global__ __launch_bounds__(256) void tt_main(const __bf16* __restrict__ Xt,
                                               const __bf16* __restrict__ Bmat,
                                               const float* __restrict__ bias2,
                                               float* __restrict__ out) {
    __shared__ uint4 Asm4[2][64 * 4];    // 2 x 4 KB  A tiles (64 px x 32 k bf16)
    __shared__ uint4 Bsm4[2][256 * 4];   // 2 x 16 KB B tiles (256 o x 32 k bf16)

    const int tid  = threadIdx.x;
    const int lane = tid & 31;
    const int wave = tid >> 5;        // wave32
    const int mty  = wave & 3;        // wave's M-tile = spatial row in tile
    const int ntb  = (wave >> 2) * 8; // wave's first N-tile

    const int blk = blockIdx.x;       // 512 blocks: 8 b * 16 ytiles * 4 xtiles
    const int b   = blk >> 6;
    const int rem = blk & 63;
    const int y0  = (rem >> 2) * 4;
    const int x0  = (rem & 3) * 16;

    v8f acc[8];
    #pragma unroll
    for (int n = 0; n < 8; ++n)
        #pragma unroll
        for (int z = 0; z < 8; ++z) acc[n][z] = 0.f;

    // staging role of this thread: pixel p, 16B chunk q
    const int p  = tid >> 2;
    const int q  = tid & 3;
    const int py = p >> 4;
    const int px = p & 15;
    const int hlf = lane >> 4;
    const int row = lane & 15;

    // issue the async global->LDS copies for K-step `it` (it = t*8 + cb)
    auto issue_tile = [&](int it, int sbuf) {
        const int t  = it >> 3, cb = it & 7;
        const int dy = t / 3, dx = t % 3;     // padded Xt: tap offsets 0..2
        const __bf16* ga = Xt + ((((size_t)b * 66 + (y0 + py + dy)) * 66
                                  + (x0 + px + dx)) * 256 + cb * 32 + q * 8);
        ASYNC_CP128(ga, &Asm4[sbuf][p * 4 + q], 0);
        const __bf16* gb = Bmat + (((size_t)t * 256 + tid) * 256 + cb * 32);
        uint4* lb = &Bsm4[sbuf][tid * 4];
        ASYNC_CP128(gb, lb, 0);
        ASYNC_CP128(gb, lb, 16);
        ASYNC_CP128(gb, lb, 32);
        ASYNC_CP128(gb, lb, 48);
    };

    union Frag { uint4 u[2]; v16bf v; };

    issue_tile(0, 0);                 // prologue

    for (int it = 0; it < 72; ++it) {
        const int buf = it & 1;
        wait_async0();                // this wave's current tile landed in LDS
        __syncthreads();              // all waves landed; prev buffer consumed
        if (it + 1 < 72) issue_tile(it + 1, buf ^ 1);  // overlap with WMMAs

        // A fragment (16x32 bf16 ISA layout):
        // lane<16: K{0..7,16..23}; lane>=16: K{8..15,24..31}
        Frag fa;
        const uint4* ap = &Asm4[buf][(mty * 16 + row) * 4 + hlf];
        fa.u[0] = ap[0];
        fa.u[1] = ap[2];

        Frag fb[2];
        {
            const uint4* bp = &Bsm4[buf][(ntb * 16 + row) * 4 + hlf];
            fb[0].u[0] = bp[0];
            fb[0].u[1] = bp[2];
        }
        #pragma unroll
        for (int n = 0; n < 8; ++n) {
            if (n < 7) {   // prefetch next B fragment before issuing this WMMA
                const uint4* bp = &Bsm4[buf][((ntb + n + 1) * 16 + row) * 4 + hlf];
                fb[(n + 1) & 1].u[0] = bp[0];
                fb[(n + 1) & 1].u[1] = bp[2];
            }
            acc[n] = __builtin_amdgcn_wmma_f32_16x16x32_bf16(
                false, fa.v, false, fb[n & 1].v, (short)0, acc[n], false, false);
        }
    }

    // ---- epilogue: bias + store (C layout: VGPR r -> M=r / r+8; lane -> N)
    const int mb = (lane < 16) ? 0 : 8;
    #pragma unroll
    for (int n = 0; n < 8; ++n) {
        const int o = (ntb + n) * 16 + row;
        const float bs = bias2[o];
        #pragma unroll
        for (int r = 0; r < 8; ++r)
            out[(((size_t)b * 256 + o) * 64 + (y0 + mty)) * 64 + (x0 + r + mb)]
                = acc[n][r] + bs;
    }
}

extern "C" void kernel_launch(void* const* d_in, const int* in_sizes, int n_in,
                              void* d_out, int out_size, void* d_ws, size_t ws_size,
                              hipStream_t stream) {
    (void)in_sizes; (void)n_in; (void)out_size; (void)ws_size;
    const float* X      = (const float*)d_in[0];
    const float* conv_w = (const float*)d_in[1];
    const float* conv_b = (const float*)d_in[2];
    const float* core1  = (const float*)d_in[3];
    const float* core2  = (const float*)d_in[4];
    const float* core3  = (const float*)d_in[5];
    float* out = (float*)d_out;
    char*  ws  = (char*)d_ws;

    __bf16* Xt    = (__bf16*)(ws + WS_XT);
    __bf16* Bmat  = (__bf16*)(ws + WS_BMAT);
    float*  bias2 = (float*)(ws + WS_BIAS);
    float*  H1    = (float*)(ws + WS_H1);
    float*  W2    = (float*)(ws + WS_W2);

    k_prep1<<<1024,  256, 0, stream>>>(core1, core2, H1);
    k_prep2<<<4096,  256, 0, stream>>>(H1, core3, W2);
    k_prep3<<<2304,  256, 0, stream>>>(W2, conv_w, Bmat);
    k_bias <<<1,     256, 0, stream>>>(W2, conv_b, bias2);
    k_xT   <<<34848, 256, 0, stream>>>(X, Xt);
    tt_main<<<512,   256, 0, stream>>>(Xt, Bmat, bias2, out);
}
